// Cross_PCLEMA_87668872446336
// MI455X (gfx1250) — matrix operand order repaired and checked by reading, not compiled, not run on hardware
//
#include <hip/hip_runtime.h>
#include <hip/hip_bf16.h>
#include <math.h>

// ---------------------------------------------------------------------------
// Cross-modal PCL + EMA codebook, fused for MI455X (gfx1250, wave32, WMMA).
//
// Shapes: N = 32*512 = 16384 rows, D = 256, M = 1024 codes.
// Heavy work = 4 GEMMs [N,256]x[256,M] streamed tile-by-tile through
// v_wmma_f32_16x16x32_bf16 with online argmin / softmax-entropy / logsumexp
// per row, so the [N,M] matrices are never materialized (memory-bound at
// ~300MB of HBM traffic -> ~13us at 23.3 TB/s; WMMA time is negligible;
// the 1MB codebook stays hot in the 192MB L2 across all 256 workgroups).
// The InfoNCE pass stages its bf16 codebook tiles with the Tensor Data
// Mover (TENSOR_LOAD_TO_LDS + s_wait_tensorcnt) instead of VALU copies.
// ---------------------------------------------------------------------------

typedef __attribute__((ext_vector_type(16))) __bf16 v16bf;
typedef __attribute__((ext_vector_type(8)))  float  v8f;
typedef unsigned int u32x4 __attribute__((ext_vector_type(4)));
typedef int          i32x4 __attribute__((ext_vector_type(4)));
typedef int          i32x8 __attribute__((ext_vector_type(8)));

#define NROWS 16384
#define DDIM  256
#define MCODE 1024
#define DECAYF 0.99f
#define EPSF   1e-5f
#define COMMITF 0.25f
#define INV_TEMP 10.0f
#define INV_LOGM (1.0f / 6.931471805599453f) // 1/ln(1024)

__device__ __forceinline__ float wave_sum32(float v) {
#pragma unroll
  for (int m = 16; m >= 1; m >>= 1) v += __shfl_xor(v, m, 32);
  return v;
}

// --- WMMA fragment loaders from row-major bf16 LDS tiles (row stride 256) ---
// A (16x32, rows = X rows): lane%16 = row; lanes 0-15 hold K {0..7,16..23},
// lanes 16-31 hold K {8..15,24..31} (per CDNA5 ISA 16-bit A layout).
__device__ __forceinline__ v16bf load_fragA(const __bf16* base, int lane, int kbase) {
  const __bf16* p = base + (lane & 15) * DDIM + kbase + ((lane & 16) ? 8 : 0);
  v16bf f;
#pragma unroll
  for (int i = 0; i < 8; ++i) { f[i] = p[i]; f[i + 8] = p[16 + i]; }
  return f;
}
// B (32x16, cols = codebook rows): lane%16 = col; lanes 0-15 K {0..7,16..23},
// lanes 16-31 K {8..15,24..31} (same K order as A).
__device__ __forceinline__ v16bf load_fragB(const __bf16* base, int lane, int kbase) {
  const __bf16* p = base + (lane & 15) * DDIM + kbase + ((lane & 16) ? 8 : 0);
  v16bf f;
#pragma unroll
  for (int i = 0; i < 8; ++i) { f[i] = p[i]; f[i + 8] = p[16 + i]; }
  return f;
}

// Issue a TDM 2D tile load: `rows16` rows x 512 bytes (128 x 4B units) from
// `gptr` (row stride 512B) into LDS at `lds_byte_addr`. Caller must
// s_wait_tensorcnt + barrier before consuming. EXEC-independent; issue from
// one wave only.
__device__ __forceinline__ void tdm_load_tile(const void* gptr, unsigned lds_byte_addr) {
  unsigned long long ga = (unsigned long long)gptr;
  u32x4 g0;
  g0[0] = 1u;                                   // count=1, user descriptor
  g0[1] = lds_byte_addr;                        // lds_addr
  g0[2] = (unsigned)(ga & 0xFFFFFFFFu);         // global_addr[31:0]
  g0[3] = (unsigned)((ga >> 32) & 0x1FFFFFFu)   // global_addr[56:32]
        | (2u << 30);                           // type=2 ("image")
  const unsigned td0 = 128, td1 = 16;           // tensor dims (4B units, rows)
  const unsigned ti0 = 128, ti1 = 16;           // tile dims
  const unsigned long long st0 = 128;           // dim0 stride (4B units)
  i32x8 g1;
  g1[0] = (int)(2u << 16);                      // mask=0, data_size=2 (4B)
  g1[1] = (int)((td0 & 0xFFFFu) << 16);         // [31:16]=tensor_dim0 lo16
  g1[2] = (int)(((td0 >> 16) & 0xFFFFu) | ((td1 & 0xFFFFu) << 16));
  g1[3] = (int)(((td1 >> 16) & 0xFFFFu) | ((ti0 & 0xFFFFu) << 16));
  g1[4] = (int)(ti1 & 0xFFFFu);                 // tile_dim1 (tile_dim2=0)
  g1[5] = (int)(st0 & 0xFFFFFFFFu);             // tensor_dim0_stride[31:0]
  g1[6] = (int)((st0 >> 32) & 0xFFFFu);         // stride[47:32]; dim1_stride lo=0
  g1[7] = 0;
  i32x4 gz = {0, 0, 0, 0};
#if defined(__clang_major__) && __clang_major__ >= 23
  i32x8 gz8 = {0, 0, 0, 0, 0, 0, 0, 0};
  __builtin_amdgcn_tensor_load_to_lds(g0, g1, gz, gz, gz8, 0);
#else
  __builtin_amdgcn_tensor_load_to_lds(g0, g1, gz, gz, 0);
#endif
}

// ---------------------------------------------------------------------------
__global__ void zero2_kernel(float* p, long n, float* q, long m) {
  long t = (long)blockIdx.x * blockDim.x + threadIdx.x;
  long stride = (long)gridDim.x * blockDim.x;
  for (long i = t; i < n; i += stride) p[i] = 0.0f;
  for (long i = t; i < m; i += stride) q[i] = 0.0f;
}

// Row sums-of-squares for audio, video (N rows) and embedding (M rows).
__global__ void rowsq_kernel(const float* __restrict__ A, const float* __restrict__ V,
                             const float* __restrict__ E,
                             float* __restrict__ xsq_a, float* __restrict__ xsq_v,
                             float* __restrict__ e_sq) {
  int lane = threadIdx.x & 31, wave = threadIdx.x >> 5;
  int j = blockIdx.x * (blockDim.x >> 5) + wave;
  if (j < NROWS) {
    const float4* pa = (const float4*)(A + (long)j * DDIM);
    const float4* pv = (const float4*)(V + (long)j * DDIM);
    float sa = 0.f, sv = 0.f;
#pragma unroll
    for (int i = 0; i < 2; ++i) {
      float4 fa = pa[lane + i * 32]; sa += fa.x*fa.x + fa.y*fa.y + fa.z*fa.z + fa.w*fa.w;
      float4 fv = pv[lane + i * 32]; sv += fv.x*fv.x + fv.y*fv.y + fv.z*fv.z + fv.w*fv.w;
    }
    sa = wave_sum32(sa); sv = wave_sum32(sv);
    if (lane == 0) { xsq_a[j] = sa; xsq_v[j] = sv; }
  } else if (j < NROWS + MCODE) {
    int m = j - NROWS;
    const float4* pe = (const float4*)(E + (long)m * DDIM);
    float se = 0.f;
#pragma unroll
    for (int i = 0; i < 2; ++i) {
      float4 fe = pe[lane + i * 32]; se += fe.x*fe.x + fe.y*fe.y + fe.z*fe.z + fe.w*fe.w;
    }
    se = wave_sum32(se);
    if (lane == 0) e_sq[m] = se;
  }
}

// Streaming distance pass: per row -> argmin(dist) and entropy-adjust of
// softmax(-sqrt(max(dist,0))) over all M codes, via WMMA + online stats.
// 128 threads = 4 waves; each wave owns 16 rows; 64 rows per workgroup.
__global__ void __launch_bounds__(128)
dist_kernel(const float* __restrict__ X, const float* __restrict__ E,
            const float* __restrict__ xsq, const float* __restrict__ esq,
            float* __restrict__ adj_out, int* __restrict__ idx_out) {
  __shared__ __bf16 xs[64 * DDIM];   // 32 KB
  __shared__ __bf16 et[16 * DDIM];   //  8 KB
  const int lane = threadIdx.x & 31;
  const int wave = threadIdx.x >> 5;
  const int rowBase = blockIdx.x * 64;

  // Stage 64 X rows fp32 -> bf16 (conversion => manual staging, not TDM)
  for (int i = threadIdx.x; i < 64 * DDIM / 4; i += blockDim.x) {
    float4 f = ((const float4*)(X + (long)rowBase * DDIM))[i];
    int o = i * 4;
    xs[o] = (__bf16)f.x; xs[o+1] = (__bf16)f.y; xs[o+2] = (__bf16)f.z; xs[o+3] = (__bf16)f.w;
  }

  const int half = (lane & 16) ? 8 : 0;
  float xsq_r[8];
#pragma unroll
  for (int r = 0; r < 8; ++r) xsq_r[r] = xsq[rowBase + wave * 16 + half + r];

  float minv[8], mx[8], Zs[8], Ss[8]; int mini[8];
#pragma unroll
  for (int r = 0; r < 8; ++r) { minv[r] = 3.4e38f; mini[r] = 0; mx[r] = -3.4e38f; Zs[r] = 0.f; Ss[r] = 0.f; }

  for (int mt = 0; mt < MCODE / 16; ++mt) {
    // stage 16 codebook rows fp32 -> bf16
    for (int i = threadIdx.x; i < 16 * DDIM / 4; i += blockDim.x) {
      float4 f = ((const float4*)(E + (long)mt * 16 * DDIM))[i];
      int o = i * 4;
      et[o] = (__bf16)f.x; et[o+1] = (__bf16)f.y; et[o+2] = (__bf16)f.z; et[o+3] = (__bf16)f.w;
    }
    if (mt + 1 < MCODE / 16)
      __builtin_prefetch(E + (long)(mt + 1) * 16 * DDIM + threadIdx.x * 32, 0, 0);
    __syncthreads();

    v8f c = {};
#pragma unroll
    for (int kt = 0; kt < 8; ++kt) {
      v16bf a = load_fragA(xs + wave * 16 * DDIM, lane, kt * 32);
      v16bf b = load_fragB(et, lane, kt * 32);
      c = __builtin_amdgcn_wmma_f32_16x16x32_bf16(false, a, false, b, (short)0, c, false, false);
    }

    const int col = mt * 16 + (lane & 15);
    const float esq_l = esq[col];
#pragma unroll
    for (int r = 0; r < 8; ++r) {
      float dist = esq_l + xsq_r[r] - 2.0f * c[r];
      if (dist < minv[r]) { minv[r] = dist; mini[r] = col; }
      float s = -sqrtf(fmaxf(dist, 0.0f));
      float nm = fmaxf(mx[r], s);
      float sc = __expf(mx[r] - nm);
      float es = __expf(s - nm);
      Zs[r] = Zs[r] * sc + es;
      Ss[r] = Ss[r] * sc + es * s;
      mx[r] = nm;
    }
    __syncthreads();
  }

  // merge stats across the 16 lanes sharing each row half
#pragma unroll
  for (int r = 0; r < 8; ++r) {
#pragma unroll
    for (int d = 1; d < 16; d <<= 1) {
      float ov = __shfl_xor(minv[r], d, 32);
      int   oi = __shfl_xor(mini[r], d, 32);
      if (ov < minv[r] || (ov == minv[r] && oi < mini[r])) { minv[r] = ov; mini[r] = oi; }
      float om = __shfl_xor(mx[r], d, 32);
      float oZ = __shfl_xor(Zs[r], d, 32);
      float oS = __shfl_xor(Ss[r], d, 32);
      float nm = fmaxf(mx[r], om);
      float s1 = __expf(mx[r] - nm), s2 = __expf(om - nm);
      Zs[r] = Zs[r] * s1 + oZ * s2;
      Ss[r] = Ss[r] * s1 + oS * s2;
      mx[r] = nm;
    }
    if ((lane & 15) == 0) {
      int row = rowBase + wave * 16 + half + r;
      float H = (mx[r] + __logf(Zs[r])) - Ss[r] / Zs[r]; // entropy of softmax
      adj_out[row] = 1.0f - H * INV_LOGM;
      idx_out[row] = mini[r];
    }
  }
}

// Per-row counts scatter + three scalar sums (S_ema, S_v, S_a).
__global__ void cnt_kernel(const int* __restrict__ a_idx, const int* __restrict__ v_idx,
                           const float* __restrict__ a_adj, const float* __restrict__ v_adj,
                           const float* __restrict__ ema_count,
                           float* __restrict__ cnt_a, float* __restrict__ cnt_v,
                           float* __restrict__ sums) {
  int t = blockIdx.x * blockDim.x + threadIdx.x;
  float va = 0.f, aa = 0.f, em = 0.f;
  if (t < NROWS) {
    va = v_adj[t]; aa = a_adj[t];
    atomicAdd(&cnt_v[v_idx[t]], va);
    atomicAdd(&cnt_a[a_idx[t]], aa);
  }
  if (t < MCODE) em = ema_count[t];
  float sva = wave_sum32(va), saa = wave_sum32(aa), sem = wave_sum32(em);
  if ((threadIdx.x & 31) == 0) {
    atomicAdd(&sums[0], sem);
    atomicAdd(&sums[1], sva);
    atomicAdd(&sums[2], saa);
  }
}

// enc^T @ (a+v) scatters into [M,D] accumulators. One block per row.
__global__ void scat_kernel(const float* __restrict__ A, const float* __restrict__ V,
                            const int* __restrict__ a_idx, const int* __restrict__ v_idx,
                            const float* __restrict__ a_adj, const float* __restrict__ v_adj,
                            float* __restrict__ scat_a, float* __restrict__ scat_v) {
  int n = blockIdx.x, d = threadIdx.x;
  float s = A[(long)n * DDIM + d] + V[(long)n * DDIM + d];
  atomicAdd(&scat_v[(long)v_idx[n] * DDIM + d], v_adj[n] * s);
  atomicAdd(&scat_a[(long)a_idx[n] * DDIM + d], a_adj[n] * s);
}

// EMA finalize + row-normalized codebook (fp32 + bf16). One wave per code row.
// The count renormalization preserves the total, so n1/n2 are scalars
// derivable from S_ema/S_v/S_a without an extra global reduction round-trip.
__global__ void finalize_kernel(const float* __restrict__ ema_weight,
                                const float* __restrict__ ema_count,
                                const float* __restrict__ cnt_a, const float* __restrict__ cnt_v,
                                const float* __restrict__ sums,
                                const float* __restrict__ scat_a, const float* __restrict__ scat_v,
                                float* __restrict__ en, __bf16* __restrict__ en_bf) {
  int lane = threadIdx.x & 31, wave = threadIdx.x >> 5;
  int m = blockIdx.x * 8 + wave;
  float S_ema = sums[0], S_v = sums[1], S_a = sums[2];
  float ec1 = DECAYF * ema_count[m] + (1.0f - DECAYF) * cnt_v[m];
  float n1  = DECAYF * S_ema + (1.0f - DECAYF) * S_v;
  ec1 = (ec1 + EPSF) / (n1 + MCODE * EPSF) * n1;
  float ec2 = DECAYF * ec1 + (1.0f - DECAYF) * cnt_a[m];
  float n2  = DECAYF * n1 + (1.0f - DECAYF) * S_a;
  ec2 = (ec2 + EPSF) / (n2 + MCODE * EPSF) * n2;
  float inv_ec = 1.0f / ec2;

  float emb[8]; float nrm = 0.f;
#pragma unroll
  for (int i = 0; i < 8; ++i) {
    long d = (long)m * DDIM + lane * 8 + i;
    float ew = DECAYF * (DECAYF * ema_weight[d] + 0.5f * (1.0f - DECAYF) * scat_v[d])
             + 0.5f * (1.0f - DECAYF) * scat_a[d];
    emb[i] = ew * inv_ec;
    nrm += emb[i] * emb[i];
  }
  nrm = wave_sum32(nrm);
  float inv = 1.0f / fmaxf(sqrtf(nrm), 1e-8f);
#pragma unroll
  for (int i = 0; i < 8; ++i) {
    long d = (long)m * DDIM + lane * 8 + i;
    float v = emb[i] * inv;
    en[d] = v;
    en_bf[d] = (__bf16)v;
  }
}

// Streaming InfoNCE logits pass: per row logsumexp of (fn . en^T)/TEMP.
// Codebook tiles (already bf16, raw copy) are staged by the Tensor Data
// Mover; wave 0 issues TENSOR_LOAD_TO_LDS and waits on TENSORcnt.
__global__ void __launch_bounds__(128)
nce_kernel(const float* __restrict__ X, const __bf16* __restrict__ ENB,
           const float* __restrict__ xsq, float* __restrict__ lse_out) {
  __shared__ __bf16 xs[64 * DDIM];
  __shared__ __bf16 et[16 * DDIM];
  const int lane = threadIdx.x & 31;
  const int wave = threadIdx.x >> 5;
  const int rowBase = blockIdx.x * 64;
  // LDS byte address of et (flat LDS addresses truncate to the low 32 bits)
  const unsigned et_lds = (unsigned)(unsigned long long)(void*)et;

  // stage normalized X rows fp32 -> bf16 (scale by 1/max(||x||,1e-8))
  for (int i = threadIdx.x; i < 64 * DDIM / 4; i += blockDim.x) {
    int row = i >> 6; // (i*4)/256
    float inv = 1.0f / fmaxf(sqrtf(xsq[rowBase + row]), 1e-8f);
    float4 f = ((const float4*)(X + (long)rowBase * DDIM))[i];
    int o = i * 4;
    xs[o] = (__bf16)(f.x * inv); xs[o+1] = (__bf16)(f.y * inv);
    xs[o+2] = (__bf16)(f.z * inv); xs[o+3] = (__bf16)(f.w * inv);
  }

  const int half = (lane & 16) ? 8 : 0;
  float mx[8], Zs[8];
#pragma unroll
  for (int r = 0; r < 8; ++r) { mx[r] = -3.4e38f; Zs[r] = 0.f; }

  for (int mt = 0; mt < MCODE / 16; ++mt) {
    // TDM: async-copy 16 codebook rows (16 x 512B) into LDS
    if (wave == 0) {
      tdm_load_tile(ENB + (long)mt * 16 * DDIM, et_lds);
      __builtin_amdgcn_s_wait_tensorcnt(0);
    }
    if (mt + 1 < MCODE / 16)
      __builtin_prefetch(ENB + (long)(mt + 1) * 16 * DDIM + threadIdx.x * 64, 0, 0);
    __syncthreads();

    v8f c = {};
#pragma unroll
    for (int kt = 0; kt < 8; ++kt) {
      v16bf a = load_fragA(xs + wave * 16 * DDIM, lane, kt * 32);
      v16bf b = load_fragB(et, lane, kt * 32);
      c = __builtin_amdgcn_wmma_f32_16x16x32_bf16(false, a, false, b, (short)0, c, false, false);
    }
#pragma unroll
    for (int r = 0; r < 8; ++r) {
      float s = c[r] * INV_TEMP;
      float nm = fmaxf(mx[r], s);
      Zs[r] = Zs[r] * __expf(mx[r] - nm) + __expf(s - nm);
      mx[r] = nm;
    }
    __syncthreads();
  }

#pragma unroll
  for (int r = 0; r < 8; ++r) {
#pragma unroll
    for (int d = 1; d < 16; d <<= 1) {
      float om = __shfl_xor(mx[r], d, 32);
      float oZ = __shfl_xor(Zs[r], d, 32);
      float nm = fmaxf(mx[r], om);
      Zs[r] = Zs[r] * __expf(mx[r] - nm) + oZ * __expf(om - nm);
      mx[r] = nm;
    }
    if ((lane & 15) == 0) {
      int row = rowBase + wave * 16 + half + r;
      lse_out[row] = mx[r] + __logf(Zs[r]);
    }
  }
}

// Final loss: gathered logits at the 4 one-hot positions + logsumexp terms.
__global__ void loss_kernel(const float* __restrict__ A, const float* __restrict__ V,
                            const float* __restrict__ en,
                            const int* __restrict__ a_idx, const int* __restrict__ v_idx,
                            const float* __restrict__ xsq_a, const float* __restrict__ xsq_v,
                            const float* __restrict__ lse_a, const float* __restrict__ lse_v,
                            float* __restrict__ out) {
  int lane = threadIdx.x & 31, wave = threadIdx.x >> 5;
  int n = blockIdx.x * 8 + wave;
  int ai = a_idx[n], vi = v_idx[n];
  float inva = 1.0f / fmaxf(sqrtf(xsq_a[n]), 1e-8f);
  float invv = 1.0f / fmaxf(sqrtf(xsq_v[n]), 1e-8f);
  float das = 0.f, dac = 0.f, dvs = 0.f, dvc = 0.f;
#pragma unroll
  for (int i = 0; i < 8; ++i) {
    int d = lane * 8 + i;
    float fa = A[(long)n * DDIM + d], fv = V[(long)n * DDIM + d];
    float ea = en[(long)ai * DDIM + d], ev = en[(long)vi * DDIM + d];
    das += fa * ea; dac += fa * ev; dvs += fv * ev; dvc += fv * ea;
  }
  das = wave_sum32(das) * inva * INV_TEMP;
  dac = wave_sum32(dac) * inva * INV_TEMP;
  dvs = wave_sum32(dvs) * invv * INV_TEMP;
  dvc = wave_sum32(dvc) * invv * INV_TEMP;
  if (lane == 0) {
    float ta = lse_a[n] - (COMMITF * das + (1.0f - COMMITF) * dac);
    float tv = lse_v[n] - (COMMITF * dvs + (1.0f - COMMITF) * dvc);
    atomicAdd(out, COMMITF * (ta + tv) / ((float)NROWS * 32.0f));
  }
}

// ---------------------------------------------------------------------------
extern "C" void kernel_launch(void* const* d_in, const int* in_sizes, int n_in,
                              void* d_out, int out_size, void* d_ws, size_t ws_size,
                              hipStream_t stream) {
  (void)in_sizes; (void)n_in; (void)out_size; (void)ws_size;
  const float* A  = (const float*)d_in[0];
  const float* V  = (const float*)d_in[1];
  const float* E  = (const float*)d_in[2];
  const float* ema_count  = (const float*)d_in[3];
  const float* ema_weight = (const float*)d_in[4];
  // d_in[5] = epoch (unused by the math)
  float* out = (float*)d_out;

  const long N = NROWS, M = MCODE, MD = (long)MCODE * DDIM;
  float* w = (float*)d_ws;
  float* xsq_a = w;                 // N
  float* xsq_v = xsq_a + N;         // N
  float* e_sq  = xsq_v + N;         // M
  float* a_adj = e_sq + M;          // N
  float* v_adj = a_adj + N;         // N
  float* lse_a = v_adj + N;         // N
  float* lse_v = lse_a + N;         // N
  int*   a_idx = (int*)(lse_v + N); // N
  int*   v_idx = a_idx + N;         // N
  float* cnt_a = (float*)(v_idx + N); // M   -- zero region begins here
  float* cnt_v = cnt_a + M;           // M
  float* sums  = cnt_v + M;           // 16
  float* scat_a = sums + 16;          // M*D
  float* scat_v = scat_a + MD;        // M*D -- zero region ends here
  float* en     = scat_v + MD;        // M*D
  __bf16* en_bf = (__bf16*)(en + MD); // M*D bf16  (~4.2 MB total workspace)

  long zcount = 2 * M + 16 + 2 * MD;
  zero2_kernel<<<2048, 256, 0, stream>>>(cnt_a, zcount, out, 1);

  rowsq_kernel<<<(NROWS + MCODE) / 8, 256, 0, stream>>>(A, V, E, xsq_a, xsq_v, e_sq);

  dist_kernel<<<NROWS / 64, 128, 0, stream>>>(A, E, xsq_a, e_sq, a_adj, a_idx);
  dist_kernel<<<NROWS / 64, 128, 0, stream>>>(V, E, xsq_v, e_sq, v_adj, v_idx);

  cnt_kernel<<<NROWS / 256, 256, 0, stream>>>(a_idx, v_idx, a_adj, v_adj, ema_count,
                                              cnt_a, cnt_v, sums);
  scat_kernel<<<NROWS, DDIM, 0, stream>>>(A, V, a_idx, v_idx, a_adj, v_adj, scat_a, scat_v);

  finalize_kernel<<<MCODE / 8, 256, 0, stream>>>(ema_weight, ema_count, cnt_a, cnt_v,
                                                 sums, scat_a, scat_v, en, en_bf);

  nce_kernel<<<NROWS / 64, 128, 0, stream>>>(A, en_bf, xsq_a, lse_a);
  nce_kernel<<<NROWS / 64, 128, 0, stream>>>(V, en_bf, xsq_v, lse_v);

  loss_kernel<<<NROWS / 8, 256, 0, stream>>>(A, V, en, a_idx, v_idx,
                                             xsq_a, xsq_v, lse_a, lse_v, out);
}
